// InstanceAttentionModule_81415400063330
// MI455X (gfx1250) — matrix operand order repaired and checked by reading, not compiled, not run on hardware
//
#include <hip/hip_runtime.h>
#include <hip/hip_bf16.h>

// Problem sizes (fixed by the reference): b=4, h=w=64 -> hw=4096, c=256, fc=64.
#define BATCH 4
#define HW    4096
#define CIN   256
#define FCH   64

typedef __bf16 bf16_t;
typedef bf16_t bf16x8  __attribute__((ext_vector_type(8)));
typedef bf16_t bf16x16 __attribute__((ext_vector_type(16)));
typedef float  f32x8   __attribute__((ext_vector_type(8)));

#define HAS_TDM         __has_builtin(__builtin_amdgcn_tensor_load_to_lds)
#define HAS_ASYNC       __has_builtin(__builtin_amdgcn_global_load_async_to_lds_b128)
#define HAS_WAIT_TENSOR __has_builtin(__builtin_amdgcn_s_wait_tensorcnt)
#define HAS_WAIT_ASYNC  __has_builtin(__builtin_amdgcn_s_wait_asynccnt)

// LDS row stride for staged P tiles: 256 elems + 8 pad elems (16B) so that the
// B-operand 16B/lane reads hit each of the 64 LDS banks exactly twice (minimum).
#define PROW 264

// Build a per-lane 16-element bf16 vector from two 16-byte chunks.
static __device__ __forceinline__ bf16x16 ldpair(const bf16_t* p0, const bf16_t* p1) {
    bf16x8 lo = *reinterpret_cast<const bf16x8*>(p0);
    bf16x8 hi = *reinterpret_cast<const bf16x8*>(p1);
    return __builtin_shufflevector(lo, hi, 0,1,2,3,4,5,6,7,8,9,10,11,12,13,14,15);
}

static __device__ __forceinline__ f32x8 wmma_bf16(bf16x16 a, bf16x16 b, f32x8 c) {
    // D = A(16x32 bf16) * B(32x16 bf16) + C(16x16 f32)
    return __builtin_amdgcn_wmma_f32_16x16x32_bf16(false, a, false, b, (short)0, c,
                                                   false, false);
}

// ---- CDNA5 staging paths: TDM (TENSORcnt) > async-to-LDS (ASYNCcnt) > copy ----
typedef unsigned int u32x4 __attribute__((ext_vector_type(4)));
typedef int          i32x8 __attribute__((ext_vector_type(8)));
typedef int          i32x4 __attribute__((ext_vector_type(4)));
typedef int          v4i   __attribute__((ext_vector_type(4)));
typedef __attribute__((address_space(1))) v4i gv4i;   // global-AS 16B block
typedef __attribute__((address_space(3))) v4i lv4i;   // LDS-AS 16B block

#if HAS_TDM
// D#: 2D tile, 2-byte elements, 16 rows x 256 cols, global row stride HW,
// LDS padding: +4 dwords after every 128 dwords (one row) -> LDS stride PROW.
static __device__ __forceinline__ void tdm_load_tile(const bf16_t* gsrc,
                                                     unsigned int lds_byte) {
    const unsigned long long ga = (unsigned long long)(uintptr_t)gsrc;
    u32x4 g0;
    g0[0] = 1u;                                        // count=1 (valid user D#)
    g0[1] = lds_byte;                                  // lds_addr
    g0[2] = (unsigned int)ga;                          // global_addr[31:0]
    g0[3] = (unsigned int)(ga >> 32) | (2u << 30);     // addr[56:32] | type=2
    i32x8 g1;
    g1[0] = (int)((1u << 16)        // data_size = 2 bytes
                | (1u << 20)        // pad_enable
                | (6u << 22)        // pad_interval: 128 dwords
                | (3u << 25));      // pad_amount:   4 dwords
    g1[1] = (int)(4096u << 16);     // tensor_dim0 = 4096 (bits 79:48, lo part)
    g1[2] = (int)(16u << 16);       // tensor_dim0 hi=0 | tensor_dim1 = 16
    g1[3] = (int)(256u << 16);      // tensor_dim1 hi=0 | tile_dim0 = 256
    g1[4] = 16;                     // tile_dim1 = 16 | tile_dim2 = 0
    g1[5] = HW;                     // tensor_dim0_stride lo32
    g1[6] = 0;                      // stride0 hi | tensor_dim1_stride lo
    g1[7] = 0;
    const i32x4 z4 = {0, 0, 0, 0};
    const i32x8 z8 = {0, 0, 0, 0, 0, 0, 0, 0};
    __builtin_amdgcn_tensor_load_to_lds(g0, g1, z4, z4, z8, 0);
}
#endif

static __device__ __forceinline__ void stage16_async(const bf16_t* g, bf16_t* l) {
#if HAS_ASYNC
    __builtin_amdgcn_global_load_async_to_lds_b128(
        (gv4i*)(uintptr_t)g,
        (lv4i*)(unsigned int)(uintptr_t)l, 0, 0);
#else
    *reinterpret_cast<bf16x8*>(l) = *reinterpret_cast<const bf16x8*>(g);
#endif
}

// Stage one 16x256 bf16 chunk of P (rows i0..i0+15, cols k0..k0+255) into lbuf.
static __device__ __forceinline__ void stage_chunk(const bf16_t* gtile, int k0,
                                                   bf16_t* lbuf, int t, int w) {
#if HAS_TDM
    if (w == 0) tdm_load_tile(gtile + k0, (unsigned int)(uintptr_t)lbuf);
    (void)t;
#else
    const int r = t >> 4, m = t & 15;              // 256 threads x 32B each
    const bf16_t* g = gtile + (size_t)r * HW + k0 + m * 16;
    bf16_t* l = lbuf + r * PROW + m * 16;
    stage16_async(g, l);
    stage16_async(g + 8, l + 8);
#endif
}

static __device__ __forceinline__ void wait_stage(int w) {
#if HAS_TDM
    if (w == 0) {
#if HAS_WAIT_TENSOR
        __builtin_amdgcn_s_wait_tensorcnt(0);
#else
        asm volatile("s_wait_tensorcnt 0" ::: "memory");
#endif
    }
#elif HAS_ASYNC
#if HAS_WAIT_ASYNC
    __builtin_amdgcn_s_wait_asynccnt(0);
#else
    asm volatile("s_wait_asynccnt 0" ::: "memory");
#endif
    (void)w;
#else
    (void)w;   // sync copies are ordered by the barrier's implicit waits
#endif
}

// ---------------------------------------------------------------------------
// K1: per 16-row tile of spatial positions:
//   scT[(b*C+c)*HW + j] = fm[b,j,:] . W_att[c,:] + b_att[c]   (f32, j-contig)
//   emb_bf[(b*HW+j)*64 + f] = bf16(fm[b,j,:] . W_emb[f,:] + b_emb[f])
//   sqn[b*HW+j] = sum_f emb^2
// ---------------------------------------------------------------------------
__global__ __launch_bounds__(256) void prep_kernel(
    const float* __restrict__ fm,
    const float* __restrict__ W_emb, const float* __restrict__ b_emb,
    const float* __restrict__ W_att, const float* __restrict__ b_att,
    bf16_t* __restrict__ emb_bf, float* __restrict__ scT,
    float* __restrict__ sqn)
{
    __shared__ float fm_s[16][CIN];
    __shared__ float emb_s[16][FCH + 1];
    const int t  = threadIdx.x;
    const int b  = blockIdx.y;
    const int j0 = blockIdx.x * 16;

    const float* fmb = fm + ((size_t)b * HW + j0) * CIN;
    for (int idx = t; idx < 16 * CIN; idx += 256)
        fm_s[idx >> 8][idx & 255] = fmb[idx];
    __syncthreads();

    {   // shortcut channel c = t (stored transposed: channel-major, j contiguous)
        const int c = t;
        float acc[16];
        const float bias = b_att[c];
#pragma unroll
        for (int jj = 0; jj < 16; ++jj) acc[jj] = bias;
        const float* wr = W_att + (size_t)c * CIN;
        for (int k = 0; k < CIN; ++k) {
            const float w = wr[k];
#pragma unroll
            for (int jj = 0; jj < 16; ++jj) acc[jj] += fm_s[jj][k] * w;
        }
        float* dst = scT + ((size_t)b * CIN + c) * HW + j0;
#pragma unroll
        for (int jj = 0; jj < 16; ++jj) dst[jj] = acc[jj];
    }

    if (t < FCH) {   // embedding channel f = t
        const int f = t;
        float acc[16];
        const float bias = b_emb[f];
#pragma unroll
        for (int jj = 0; jj < 16; ++jj) acc[jj] = bias;
        const float* wr = W_emb + (size_t)f * CIN;
        for (int k = 0; k < CIN; ++k) {
            const float w = wr[k];
#pragma unroll
            for (int jj = 0; jj < 16; ++jj) acc[jj] += fm_s[jj][k] * w;
        }
#pragma unroll
        for (int jj = 0; jj < 16; ++jj) {
            emb_s[jj][f] = acc[jj];
            emb_bf[((size_t)b * HW + j0 + jj) * FCH + f] = (bf16_t)acc[jj];
        }
    }
    __syncthreads();
    if (t < 16) {
        float s = 0.f;
        for (int k = 0; k < FCH; ++k) { const float e = emb_s[t][k]; s += e * e; }
        sqn[(size_t)b * HW + j0 + t] = s;
    }
}

// ---------------------------------------------------------------------------
// K2: T[b*HW+j] = sum_c scT[c][j] * W_thr[c] + b_thr
// ---------------------------------------------------------------------------
__global__ __launch_bounds__(256) void thr_kernel(
    const float* __restrict__ scT, const float* __restrict__ W_thr,
    const float* __restrict__ b_thr, float* __restrict__ T)
{
    const int id = blockIdx.x * 256 + threadIdx.x;   // 0 .. B*HW-1
    const int b = id >> 12, j = id & (HW - 1);
    float acc = b_thr[0];
    const float* p = scT + (size_t)b * CIN * HW + j;
    for (int c = 0; c < CIN; ++c) acc += p[(size_t)c * HW] * W_thr[c];
    T[id] = acc;
}

// ---------------------------------------------------------------------------
// K3: scores. Block = one 16-wide i-tile; 8 waves sweep all j-tiles.
//   Gram tile via bf16 WMMA with M=j, N=i so each lane's 8 D values are 8
//   consecutive j for one i -> single 16B bf16 store into row-major P.
//   P[b,i,j] = exp(relu(exp(-relu(qi+qj-2G)) - T[j]))
//   Per-i-tile column partial sums S_part (deterministic, no float atomics).
// ---------------------------------------------------------------------------
__global__ __launch_bounds__(256) void score_kernel(
    const bf16_t* __restrict__ emb_bf, const float* __restrict__ sqn,
    const float* __restrict__ T, bf16_t* __restrict__ P,
    float* __restrict__ S_part)
{
    __shared__ float S_lds[HW];
    __shared__ float T_lds[HW];
    __shared__ float q_lds[HW];

    const int t  = threadIdx.x;
    const int b  = blockIdx.y;
    const int i0 = blockIdx.x * 16;

    for (int idx = t; idx < HW; idx += 256) {
        S_lds[idx] = 0.f;
        T_lds[idx] = T[(size_t)b * HW + idx];
        q_lds[idx] = sqn[(size_t)b * HW + idx];
    }
    __syncthreads();

    const int w = t >> 5, lane = t & 31, nl = lane & 15, h = lane >> 4;
    const bf16_t* eb = emb_bf + (size_t)b * HW * FCH;

    const bf16_t* bi = eb + (size_t)(i0 + nl) * FCH + 16 * h;
    const bf16x16 Bv0 = ldpair(bi, bi + 8);           // K 0..31
    const bf16x16 Bv1 = ldpair(bi + 32, bi + 40);     // K 32..63
    const float qi = q_lds[i0 + nl];

    for (int it = 0; it < 32; ++it) {
        const int j0 = (w + 8 * it) * 16;             // wave-exclusive j tiles
        const bf16_t* aj = eb + (size_t)(j0 + nl) * FCH + 16 * h;
        const bf16x16 A0 = ldpair(aj, aj + 8);        // K 0..31
        const bf16x16 A1 = ldpair(aj + 32, aj + 40);  // K 32..63

        f32x8 acc = {0.f, 0.f, 0.f, 0.f, 0.f, 0.f, 0.f, 0.f};
        acc = wmma_bf16(A0, Bv0, acc);
        acc = wmma_bf16(A1, Bv1, acc);

        bf16x8 pv;
        float  pr[8];
#pragma unroll
        for (int v = 0; v < 8; ++v) {
            const int j = j0 + 8 * h + v;
            float sqd = qi + q_lds[j] - 2.0f * acc[v];
            sqd = fmaxf(sqd, 0.f);
            const float sim = __expf(-sqd);
            const float a   = fmaxf(sim - T_lds[j], 0.f);
            const float p   = __expf(a);
            pr[v] = p;
            pv[v] = (bf16_t)p;
        }
        *reinterpret_cast<bf16x8*>(
            P + ((size_t)b * HW + (i0 + nl)) * HW + j0 + 8 * h) = pv;

#pragma unroll
        for (int v = 0; v < 8; ++v) {
            float r = pr[v];
            r += __shfl_xor(r, 1, 32);
            r += __shfl_xor(r, 2, 32);
            r += __shfl_xor(r, 4, 32);
            r += __shfl_xor(r, 8, 32);
            if (nl == 0) S_lds[j0 + 8 * h + v] += r;   // wave-exclusive j range
        }
    }
    __syncthreads();
    float* sp = S_part + ((size_t)b * (HW / 16) + blockIdx.x) * HW;
    for (int idx = t; idx < HW; idx += 256) sp[idx] = S_lds[idx];
}

// ---------------------------------------------------------------------------
// K4: Sinv[b,j] = 1 / sum_tiles S_part   (deterministic reduction order)
// ---------------------------------------------------------------------------
__global__ __launch_bounds__(256) void sreduce_kernel(
    const float* __restrict__ S_part, float* __restrict__ Sinv)
{
    const int id = blockIdx.x * 256 + threadIdx.x;   // 0 .. B*HW-1
    const int b = id >> 12, j = id & (HW - 1);
    const float* p = S_part + (size_t)b * (HW / 16) * HW + j;
    float s = 0.f;
    for (int tt = 0; tt < HW / 16; ++tt) s += p[(size_t)tt * HW];
    Sinv[id] = 1.0f / s;
}

// ---------------------------------------------------------------------------
// K5: scT_bf[e] = bf16(scT[e] * Sinv[b,j])   (folds softmax denom into A)
// ---------------------------------------------------------------------------
__global__ __launch_bounds__(256) void scale_kernel(
    const float* __restrict__ scT, const float* __restrict__ Sinv,
    bf16_t* __restrict__ scT_bf)
{
    const size_t e = (size_t)blockIdx.x * 256 + threadIdx.x;  // B*C*HW elems
    const int j = (int)(e & (HW - 1));
    const int b = (int)(e >> 20);                              // C*HW = 2^20
    scT_bf[e] = (bf16_t)(scT[e] * Sinv[(b << 12) + j]);
}

// ---------------------------------------------------------------------------
// K6: inst GEMM. D(m=c, n=i) = sum_j scT_bf[c][j] * P[i][j]; out = fm + D.
//   The shared B operand (P rows of this block's i-tile) is staged into LDS
//   once per 256-wide K-chunk via TDM / async-to-LDS, double buffered, then
//   read back with conflict-minimal ds_load_b128. 16 WMMA per chunk body.
// ---------------------------------------------------------------------------
__global__ __launch_bounds__(256) void inst_kernel(
    const bf16_t* __restrict__ P, const bf16_t* __restrict__ scT_bf,
    const float* __restrict__ fm, float* __restrict__ out)
{
    __shared__ bf16_t Pb[2][16 * PROW];   // 2 x 8.25KB staged P chunks

    const int t  = threadIdx.x;
    const int b  = blockIdx.y;
    const int i0 = blockIdx.x * 16;
    const int w = t >> 5, lane = t & 31, nl = lane & 15, h = lane >> 4;

    const bf16_t* pbase = P + ((size_t)b * HW + i0) * HW;      // tile rows
    const bf16_t* sbase = scT_bf + (size_t)b * CIN * HW;
    const int c0 = w * 32;
    const bf16_t* a0row = sbase + (size_t)(c0 + nl) * HW;      // c-tile 0
    const bf16_t* a1row = sbase + (size_t)(c0 + 16 + nl) * HW; // c-tile 1

    f32x8 acc0 = {0.f, 0.f, 0.f, 0.f, 0.f, 0.f, 0.f, 0.f};
    f32x8 acc1 = {0.f, 0.f, 0.f, 0.f, 0.f, 0.f, 0.f, 0.f};

    stage_chunk(pbase, 0, &Pb[0][0], t, w);        // prologue: chunk 0 -> buf 0

    for (int kb = 0; kb < 16; ++kb) {
        const int cur = kb & 1;
        wait_stage(w);
        __syncthreads();                            // Pb[cur] published
        if (kb + 1 < 16)
            stage_chunk(pbase, (kb + 1) * 256, &Pb[cur ^ 1][0], t, w);

        const bf16_t* LB = &Pb[cur][0];
        const int kg = kb * 256;
#pragma unroll
        for (int s = 0; s < 8; ++s) {
            const int kl = 32 * s;
            const bf16x16 Bv = ldpair(LB + nl * PROW + kl + 16 * h,
                                      LB + nl * PROW + kl + 16 * h + 8);
            const int k0 = kg + kl;
            __builtin_prefetch(a0row + k0 + 512, 0, 1);
            const bf16x16 A0 = ldpair(a0row + k0 + 16 * h, a0row + k0 + 16 * h + 8);
            const bf16x16 A1 = ldpair(a1row + k0 + 16 * h, a1row + k0 + 16 * h + 8);
            acc0 = wmma_bf16(A0, Bv, acc0);
            acc1 = wmma_bf16(A1, Bv, acc1);
        }
        __syncthreads();                            // done reading Pb[cur]
    }

    const int i = i0 + nl;
    const float* fr = fm + ((size_t)b * HW + i) * CIN;
    float* orow     = out + ((size_t)b * HW + i) * CIN;
    const int ca = c0 + 8 * h;
    const int cb = c0 + 16 + 8 * h;
#pragma unroll
    for (int v = 0; v < 8; ++v) {
        orow[ca + v] = fr[ca + v] + acc0[v];
        orow[cb + v] = fr[cb + v] + acc1[v];
    }
}

// ---------------------------------------------------------------------------
// Launch. Workspace layout (bytes, regions naturally aligned):
//   emb_bf 2MB | scT 16.8MB | scT_bf 8.4MB | sqn/T/Sinv 64KB each
//   S_part 16.8MB | P 134.2MB  => ~178.5MB total in d_ws
// ---------------------------------------------------------------------------
extern "C" void kernel_launch(void* const* d_in, const int* in_sizes, int n_in,
                              void* d_out, int out_size, void* d_ws, size_t ws_size,
                              hipStream_t stream) {
    const float* fm    = (const float*)d_in[0];
    const float* W_emb = (const float*)d_in[1];
    const float* b_emb = (const float*)d_in[2];
    const float* W_att = (const float*)d_in[3];
    const float* b_att = (const float*)d_in[4];
    const float* W_thr = (const float*)d_in[5];
    const float* b_thr = (const float*)d_in[6];
    float* out = (float*)d_out;

    char* ws = (char*)d_ws;
    size_t off = 0;
    bf16_t* emb_bf = (bf16_t*)(ws + off); off += (size_t)BATCH * HW * FCH * 2;
    float*  scT    = (float*) (ws + off); off += (size_t)BATCH * CIN * HW * 4;
    bf16_t* scT_bf = (bf16_t*)(ws + off); off += (size_t)BATCH * CIN * HW * 2;
    float*  sqn    = (float*) (ws + off); off += (size_t)BATCH * HW * 4;
    float*  T      = (float*) (ws + off); off += (size_t)BATCH * HW * 4;
    float*  Sinv   = (float*) (ws + off); off += (size_t)BATCH * HW * 4;
    float*  S_part = (float*) (ws + off); off += (size_t)BATCH * (HW / 16) * HW * 4;
    bf16_t* P      = (bf16_t*)(ws + off); off += (size_t)BATCH * HW * HW * 2;

    prep_kernel<<<dim3(HW / 16, BATCH), 256, 0, stream>>>(
        fm, W_emb, b_emb, W_att, b_att, emb_bf, scT, sqn);
    thr_kernel<<<(BATCH * HW) / 256, 256, 0, stream>>>(scT, W_thr, b_thr, T);
    score_kernel<<<dim3(HW / 16, BATCH), 256, 0, stream>>>(emb_bf, sqn, T, P, S_part);
    sreduce_kernel<<<(BATCH * HW) / 256, 256, 0, stream>>>(S_part, Sinv);
    scale_kernel<<<(int)(((size_t)BATCH * CIN * HW) / 256), 256, 0, stream>>>(
        scT, Sinv, scT_bf);
    inst_kernel<<<dim3(HW / 16, BATCH), 256, 0, stream>>>(P, scT_bf, fm, out);
}
